// ScaledDotProudct_73727408603350
// MI455X (gfx1250) — matrix operand order repaired and checked by reading, not compiled, not run on hardware
//
#include <hip/hip_runtime.h>
#include <hip/hip_bf16.h>

// Problem constants (from reference): B=8, S=2048, D=1024
#define BATCH 8
#define SEQ   2048
#define DIM   1024
#define SCALE 0.03125f   // 1/sqrt(1024)

// GEMM tiling
#define TM 128          // workgroup tile M
#define TN 128          // workgroup tile N
#define KC 64           // K elements staged per barrier (2 WMMA K-steps)
#define LDF 72          // padded LDS row stride in f16 (144B = 9*16B; gcd(36dw,64banks)=4)

typedef __attribute__((ext_vector_type(16))) _Float16 v16h;
typedef __attribute__((ext_vector_type(8)))  _Float16 v8h;
typedef __attribute__((ext_vector_type(4)))  _Float16 v4h;
typedef __attribute__((ext_vector_type(8)))  float    v8f;

// Build A fragment (16x32 f16): lane half h -> K = {h*16 .. h*16+15}
__device__ __forceinline__ v16h load_a_frag(const _Float16* row, int half) {
    v8h lo = *(const v8h*)(row + half * 16);
    v8h hi = *(const v8h*)(row + half * 16 + 8);
    v16h a;
#pragma unroll
    for (int t = 0; t < 8; ++t) { a[t] = lo[t]; a[t + 8] = hi[t]; }
    return a;
}

// Build B fragment (32x16 f16): lane half h -> K = {h*16 .. h*16+15}
__device__ __forceinline__ v16h load_b_frag(const _Float16* row, int half) {
    v8h lo = *(const v8h*)(row + half * 16);
    v8h hi = *(const v8h*)(row + half * 16 + 8);
    v16h b;
#pragma unroll
    for (int t = 0; t < 8; ++t) { b[t] = lo[t]; b[t + 8] = hi[t]; }
    return b;
}

// Kernel 1: WT[p][q] = (sum_s A[s,p] * Bm[s,q]) * SCALE, stored f16.
// Launched with A=V, Bm=K so WT = (V^T K)/sqrt(D) = ((K^T V)^T)/sqrt(D).
__global__ __launch_bounds__(256) void ktv_kernel(const float* __restrict__ Am,
                                                  const float* __restrict__ Bm,
                                                  _Float16* __restrict__ WT) {
    const int b  = blockIdx.z;
    const int n0 = blockIdx.x * TN;
    const int m0 = blockIdx.y * TM;
    const float* Ab = Am + (size_t)b * SEQ * DIM;
    const float* Bb = Bm + (size_t)b * SEQ * DIM;
    _Float16*    Wb = WT + (size_t)b * DIM * DIM;

    __shared__ _Float16 As[TM * LDF];
    __shared__ _Float16 Bs[TN * LDF];

    const int tid  = threadIdx.x;
    const int lane = tid & 31;
    const int wave = tid >> 5;
    const int half = lane >> 4;
    const int l16  = lane & 15;
    const int wm   = (wave & 3) * 32;   // 4 waves along M
    const int wn   = (wave >> 2) * 64;  // 2 waves along N

    v8f acc[2][4];
#pragma unroll
    for (int i = 0; i < 2; ++i)
#pragma unroll
        for (int j = 0; j < 4; ++j) acc[i][j] = (v8f){0.f,0.f,0.f,0.f,0.f,0.f,0.f,0.f};

    // loader: 256 threads cover 128 cols x 64 k (each thread: 32 k, coalesced across cols)
    const int tc  = tid & 127;
    const int tks = (tid >> 7) * 32;

    for (int k0 = 0; k0 < SEQ; k0 += KC) {
        __syncthreads();
#pragma unroll 4
        for (int kk = 0; kk < 32; ++kk) {
            const int k = tks + kk;
            const size_t grow = (size_t)(k0 + k) * DIM;
            As[tc * LDF + k] = (_Float16)Ab[grow + (m0 + tc)];   // As[p][k] (transposed stage)
            Bs[tc * LDF + k] = (_Float16)Bb[grow + (n0 + tc)];   // Bs[q][k] (transposed stage)
        }
        __syncthreads();

#pragma unroll
        for (int ks = 0; ks < 2; ++ks) {
            v16h a0 = load_a_frag(&As[(wm + l16) * LDF + ks * 32], half);
            v16h a1 = load_a_frag(&As[(wm + 16 + l16) * LDF + ks * 32], half);
#pragma unroll
            for (int j = 0; j < 4; ++j) {
                const v16h bf = load_b_frag(&Bs[(wn + j * 16 + l16) * LDF + ks * 32], half);
                acc[0][j] = __builtin_amdgcn_wmma_f32_16x16x32_f16(
                    false, a0, false, bf, (short)0, acc[0][j], false, false);
                acc[1][j] = __builtin_amdgcn_wmma_f32_16x16x32_f16(
                    false, a1, false, bf, (short)0, acc[1][j], false, false);
            }
        }
    }

    // C/D layout: VGPR r -> (m_local = r + half*8, n_local = l16); coalesced f16 stores over q
#pragma unroll
    for (int i = 0; i < 2; ++i)
#pragma unroll
        for (int j = 0; j < 4; ++j)
#pragma unroll
            for (int r = 0; r < 8; ++r) {
                const int p = m0 + wm + i * 16 + half * 8 + r;
                const int q = n0 + wn + j * 16 + l16;
                Wb[(size_t)p * DIM + q] = (_Float16)(acc[i][j][r] * SCALE);
            }
}

// Kernel 2: out_pre[b] = Q_b @ W_b where W[k][n] = WT[n][k] (WT already scaled), fp32 out.
// Dynamic LDS layout: As = smem[0 .. TM*LDF), Bs = smem[TM*LDF .. TM*LDF + TN*LDF)
#define BS_OFF (TM * LDF)
__global__ __launch_bounds__(256) void qw_kernel(const float* __restrict__ Qm,
                                                 const _Float16* __restrict__ WT,
                                                 float* __restrict__ Out) {
    extern __shared__ _Float16 smem[];
    _Float16* As = smem;
    _Float16* Bs = smem + BS_OFF;

    const int b  = blockIdx.z;
    const int n0 = blockIdx.x * TN;
    const int m0 = blockIdx.y * TM;
    const float*    Qb = Qm  + (size_t)b * SEQ * DIM;
    const _Float16* Wb = WT  + (size_t)b * DIM * DIM;
    float*          Ob = Out + (size_t)b * SEQ * DIM;

    const int tid  = threadIdx.x;
    const int lane = tid & 31;
    const int wave = tid >> 5;
    const int half = lane >> 4;
    const int l16  = lane & 15;
    const int wm   = (wave & 3) * 32;
    const int wn   = (wave >> 2) * 64;

    v8f acc[2][4];
#pragma unroll
    for (int i = 0; i < 2; ++i)
#pragma unroll
        for (int j = 0; j < 4; ++j) acc[i][j] = (v8f){0.f,0.f,0.f,0.f,0.f,0.f,0.f,0.f};

    // A loader: As[m][k] row copy of Q (fp32 -> f16). thread: row tid>>1, 32 consecutive k.
    const int am   = tid >> 1;
    const int akof = (tid & 1) * 32;
    // B loader: Bs[n][k] row copy of WT (f16), via async global->LDS b128.
    // thread: row tid>>1, 32 consecutive f16 (64B) => 4 x b128.
    const int bn   = tid >> 1;
    const int bkof = (tid & 1) * 32;

    for (int k0 = 0; k0 < DIM; k0 += KC) {
        __syncthreads();   // all waves done reading previous tiles before LDS is overwritten

        // ---- async DMA: WT row chunk -> Bs (no VGPR round trip, ASYNCcnt-tracked) ----
        {
            const _Float16* gsrc = Wb + (size_t)(n0 + bn) * DIM + (k0 + bkof);
            unsigned ldsoff = (unsigned)((BS_OFF + bn * LDF + bkof) * 2);  // bytes
#pragma unroll
            for (int i = 0; i < 4; ++i) {
                asm volatile("global_load_async_to_lds_b128 %0, %1, off"
                             :: "v"(ldsoff + i * 16), "v"(gsrc + i * 8)
                             : "memory");
            }
        }

        // ---- A stage: vectorized fp32 loads, convert, packed f16 stores ----
#pragma unroll
        for (int i = 0; i < 8; ++i) {
            const float4 f = *(const float4*)(Qb + (size_t)(m0 + am) * DIM + (k0 + akof) + i * 4);
            v4h h;
            h[0] = (_Float16)f.x; h[1] = (_Float16)f.y;
            h[2] = (_Float16)f.z; h[3] = (_Float16)f.w;
            *(v4h*)(&As[am * LDF + akof + i * 4]) = h;
        }

        asm volatile("s_wait_asynccnt 0x0" ::: "memory");
        __syncthreads();

#pragma unroll
        for (int ks = 0; ks < 2; ++ks) {
            v16h a0 = load_a_frag(&As[(wm + l16) * LDF + ks * 32], half);
            v16h a1 = load_a_frag(&As[(wm + 16 + l16) * LDF + ks * 32], half);
#pragma unroll
            for (int j = 0; j < 4; ++j) {
                const v16h bf = load_b_frag(&Bs[(wn + j * 16 + l16) * LDF + ks * 32], half);
                acc[0][j] = __builtin_amdgcn_wmma_f32_16x16x32_f16(
                    false, a0, false, bf, (short)0, acc[0][j], false, false);
                acc[1][j] = __builtin_amdgcn_wmma_f32_16x16x32_f16(
                    false, a1, false, bf, (short)0, acc[1][j], false, false);
            }
        }
    }

#pragma unroll
    for (int i = 0; i < 2; ++i)
#pragma unroll
        for (int j = 0; j < 4; ++j)
#pragma unroll
            for (int r = 0; r < 8; ++r) {
                const int m = m0 + wm + i * 16 + half * 8 + r;
                const int n = n0 + wn + j * 16 + l16;
                Ob[(size_t)m * DIM + n] = acc[i][j][r];
            }
}

// Kernel 3: row-wise softmax over last axis (D=1024), in place on d_out
__global__ __launch_bounds__(256) void softmax_kernel(float* __restrict__ Out) {
    __shared__ float red[256];
    float* p = Out + (size_t)blockIdx.x * DIM;
    const int t = threadIdx.x;

    float v[4];
    float m = -3.4e38f;
#pragma unroll
    for (int i = 0; i < 4; ++i) { v[i] = p[t + i * 256]; m = fmaxf(m, v[i]); }
    red[t] = m;
    __syncthreads();
    for (int s = 128; s > 0; s >>= 1) {
        if (t < s) red[t] = fmaxf(red[t], red[t + s]);
        __syncthreads();
    }
    m = red[0];
    __syncthreads();

    float sum = 0.f;
#pragma unroll
    for (int i = 0; i < 4; ++i) { v[i] = __expf(v[i] - m); sum += v[i]; }
    red[t] = sum;
    __syncthreads();
    for (int s = 128; s > 0; s >>= 1) {
        if (t < s) red[t] += red[t + s];
        __syncthreads();
    }
    const float inv = 1.0f / red[0];
#pragma unroll
    for (int i = 0; i < 4; ++i) p[t + i * 256] = v[i] * inv;
}

extern "C" void kernel_launch(void* const* d_in, const int* in_sizes, int n_in,
                              void* d_out, int out_size, void* d_ws, size_t ws_size,
                              hipStream_t stream) {
    const float* qm = (const float*)d_in[0];
    const float* km = (const float*)d_in[1];
    const float* vm = (const float*)d_in[2];
    float*    out = (float*)d_out;
    _Float16* W   = (_Float16*)d_ws;   // needs 8*1024*1024*2 = 16 MB

    // WT = (V^T K) * scale  (== transposed (K^T V) * scale): swap inputs, same kernel.
    dim3 g1(DIM / TN, DIM / TM, BATCH);   // 8 x 8 x 8
    ktv_kernel<<<g1, 256, 0, stream>>>(vm, km, W);

    dim3 g2(DIM / TN, SEQ / TM, BATCH);   // 8 x 16 x 8
    const size_t smem_bytes = (size_t)(TM * LDF + TN * LDF) * sizeof(_Float16);
    qw_kernel<<<g2, 256, smem_bytes, stream>>>(qm, W, out);

    softmax_kernel<<<BATCH * SEQ, 256, 0, stream>>>(out);
}
